// GAT_Net2_8323646620241
// MI455X (gfx1250) — matrix operand order repaired and checked by reading, not compiled, not run on hardware
//
#include <hip/hip_runtime.h>
#include <math.h>

// ---------------------------------------------------------------------------
// GAT 2-layer forward for MI455X (gfx1250, wave32).
//  - Dense projections via v_wmma_f32_16x16x32_f16 (fp32 accumulate).
//  - W pre-packed to f16 B-fragment layout (zero-padded) -> branch-free loop.
//  - A fragments loaded as contiguous float4 runs (global_load_b128).
//  - All NT B fragments batch-loaded per chunk -> clause + incremental waits.
//  - W_dst/W_dst2 projections collapsed against a_dst vectors (GEMM -> GEMV).
//  - Edge softmax: 3 atomic passes (ordered-uint float max, f32 atomic adds).
// ---------------------------------------------------------------------------

typedef __attribute__((ext_vector_type(16))) _Float16 v16h;
typedef __attribute__((ext_vector_type(8)))  float    v8f;

#define LRELU_SLOPE 0.2f

// Monotonic float<->uint encoding so atomicMax(u32) == float max (handles sign).
__device__ __forceinline__ unsigned encf(float f) {
  unsigned b = __float_as_uint(f);
  return (b & 0x80000000u) ? ~b : (b | 0x80000000u);
}
__device__ __forceinline__ float decf(unsigned m) {
  unsigned b = (m & 0x80000000u) ? (m ^ 0x80000000u) : ~m;
  return __uint_as_float(b);
}

// ---------------------------------------------------------------------------
// Pack W[K x ncols] (f32) into per-fragment f16 layout:
//   bp[(ch*ntile + t)*512 + lane*16 + e] = W[ch*32 + (lane>>4)*8 + (e&7) + (e>>3)*16][t*16 + (lane&15)]
// Zero-pads K and column tails so the GEMM loop needs no bounds checks.
// ---------------------------------------------------------------------------
__global__ void pack_b_k(const float* __restrict__ W, _Float16* __restrict__ bp,
                         int K, int ncols, int ntile, int nchunk) {
  int i = blockIdx.x * blockDim.x + threadIdx.x;
  int total = nchunk * ntile * 512;
  if (i >= total) return;
  int e    = i & 15;
  int lane = (i >> 4) & 31;
  int tt   = (i >> 9) % ntile;
  int ch   = (i >> 9) / ntile;
  int g = lane >> 4, n = lane & 15;
  int k   = ch * 32 + g * 8 + e + ((e < 8) ? 0 : 8);
  int col = tt * 16 + n;
  bp[i] = (k < K && col < ncols) ? (_Float16)W[(size_t)k * ncols + col] : (_Float16)0.0f;
}

__device__ __forceinline__ v16h cvt_a_frag(float4 f0, float4 f1, float4 f2, float4 f3) {
  v16h a;
  a[0]  = (_Float16)f0.x; a[1]  = (_Float16)f0.y;
  a[2]  = (_Float16)f0.z; a[3]  = (_Float16)f0.w;
  a[4]  = (_Float16)f1.x; a[5]  = (_Float16)f1.y;
  a[6]  = (_Float16)f1.z; a[7]  = (_Float16)f1.w;
  a[8]  = (_Float16)f2.x; a[9]  = (_Float16)f2.y;
  a[10] = (_Float16)f2.z; a[11] = (_Float16)f2.w;
  a[12] = (_Float16)f3.x; a[13] = (_Float16)f3.y;
  a[14] = (_Float16)f3.z; a[15] = (_Float16)f3.w;
  return a;
}

// ---------------------------------------------------------------------------
// WMMA GEMM: C[M x ncols] = A[M x K] * W[K x ncols]   (f16 inputs, f32 acc)
// One wave per 16-row block; NT 16-col tiles held in registers.
// A-fragment: lane needs k = kc+g*8..+7 and kc+16+g*8..+7 (contiguous runs).
// ---------------------------------------------------------------------------
template <int NT>
__global__ void wmma_gemm_k(const float* __restrict__ A, const _Float16* __restrict__ Bp,
                            float* __restrict__ C, int M, int K, int ncols, int ldc,
                            int nchunk) {
  const int lane = threadIdx.x & 31;
  const int mrow = lane & 15;   // M index (A) / N index (B,C)
  const int g    = lane >> 4;   // lane half
  const int row0 = blockIdx.x * 16;
  if (row0 >= M) return;
  const float* arow = A + (size_t)(row0 + mrow) * K;
  const v16h* bfrag = (const v16h*)Bp + lane;

  v8f acc[NT] = {};
  const int nfull = K >> 5;     // branch-free chunks

  for (int ch = 0; ch < nfull; ++ch) {
    const int kc = ch * 32;
    // A: two contiguous 8-float runs -> 4x b128 loads, convert in-register.
    const float4* p0 = (const float4*)(arow + kc + g * 8);
    const float4* p1 = (const float4*)(arow + kc + 16 + g * 8);
    float4 f0 = p0[0], f1 = p0[1];
    float4 f2 = p1[0], f3 = p1[1];
    v16h a = cvt_a_frag(f0, f1, f2, f3);
    // Batch all NT B-fragment loads, then run the wmma chain.
    v16h barr[NT];
#pragma unroll
    for (int t = 0; t < NT; ++t) barr[t] = bfrag[(size_t)(ch * NT + t) * 32];
#pragma unroll
    for (int t = 0; t < NT; ++t)
      acc[t] = __builtin_amdgcn_wmma_f32_16x16x32_f16(
          false, a, false, barr[t], (short)0, acc[t], false, false);
  }

  if (nfull < nchunk) {  // masked K tail (executed at most once)
    const int kc = nfull * 32;
    v16h a;
#pragma unroll
    for (int e = 0; e < 16; ++e) {
      int k = kc + g * 8 + e + ((e < 8) ? 0 : 8);
      a[e] = (_Float16)((k < K) ? arow[k] : 0.0f);
    }
    v16h barr[NT];
#pragma unroll
    for (int t = 0; t < NT; ++t) barr[t] = bfrag[(size_t)(nfull * NT + t) * 32];
#pragma unroll
    for (int t = 0; t < NT; ++t)
      acc[t] = __builtin_amdgcn_wmma_f32_16x16x32_f16(
          false, a, false, barr[t], (short)0, acc[t], false, false);
  }

#pragma unroll
  for (int t = 0; t < NT; ++t) {
    const int col = t * 16 + mrow;
    if (col < ncols) {
#pragma unroll
      for (int r = 0; r < 8; ++r) {
        int mm = row0 + r + 8 * g;  // D layout: VGPR r, lanes16-31 -> M=r+8
        C[(size_t)mm * ldc + col] = acc[t][r];
      }
    }
  }
}

// wo[f*ldo + off + h] = sum_c W[f, h*C + c] * av[h*C + c]
__global__ void collapse_k(const float* __restrict__ W, const float* __restrict__ av,
                           float* __restrict__ wo, int F, int H, int C, int ldo, int off) {
  int i = blockIdx.x * blockDim.x + threadIdx.x;
  if (i >= F * H) return;
  int f = i / H, h = i - f * H;
  float s = 0.f;
  for (int c = 0; c < C; ++c) s += W[(size_t)f * (H * C) + h * C + c] * av[h * C + c];
  wo[(size_t)f * ldo + off + h] = s;
}

// al[n*H + h] = sum_c xs[n, h*C + c] * av[h*C + c]
__global__ void als_k(const float* __restrict__ xs, const float* __restrict__ av,
                      float* __restrict__ al, int N, int H, int C) {
  int i = blockIdx.x * blockDim.x + threadIdx.x;
  if (i >= N * H) return;
  int n = i / H, h = i - n * H;
  const float* r = xs + (size_t)n * (H * C) + h * C;
  float s = 0.f;
  for (int c = 0; c < C; ++c) s += r[c] * av[h * C + c];
  al[i] = s;
}

// al[n*HW + j] = sum_f X[n,f] * wcol[f*HW + j]   (HW <= 8)
__global__ void dotal_k(const float* __restrict__ X, const float* __restrict__ wcol,
                        float* __restrict__ al, int N, int F, int HW) {
  int n = blockIdx.x * blockDim.x + threadIdx.x;
  if (n >= N) return;
  float s[8] = {0.f, 0.f, 0.f, 0.f, 0.f, 0.f, 0.f, 0.f};
  const float* xr = X + (size_t)n * F;
  for (int f = 0; f < F; ++f) {
    float xv = xr[f];
    for (int j = 0; j < HW; ++j) s[j] += xv * wcol[(size_t)f * HW + j];
  }
  for (int j = 0; j < HW; ++j) al[(size_t)n * HW + j] = s[j];
}

__global__ void fill_f(float* p, float v, size_t n) {
  size_t i = (size_t)blockIdx.x * blockDim.x + threadIdx.x;
  size_t st = (size_t)gridDim.x * blockDim.x;
  for (; i < n; i += st) p[i] = v;
}
__global__ void fill_u(unsigned* p, unsigned v, size_t n) {
  size_t i = (size_t)blockIdx.x * blockDim.x + threadIdx.x;
  size_t st = (size_t)gridDim.x * blockDim.x;
  for (; i < n; i += st) p[i] = v;
}

__device__ __forceinline__ float edge_score(const float* als, int lds_, const float* ald,
                                            int ldd, int s, int d, int h) {
  float a = als[(size_t)s * lds_ + h] + ald[(size_t)d * ldd + h];
  return (a > 0.f) ? a : LRELU_SLOPE * a;
}

__global__ void edge_amax_k(const int* __restrict__ src, const int* __restrict__ dst,
                            const float* __restrict__ als, int lds_,
                            const float* __restrict__ ald, int ldd,
                            unsigned* __restrict__ amax, int E, int H) {
  int i = blockIdx.x * blockDim.x + threadIdx.x;
  if (i >= E * H) return;
  int e = i / H, h = i - e * H;
  int s = src[e], d = dst[e];
  float a = edge_score(als, lds_, ald, ldd, s, d, h);
  atomicMax(&amax[(size_t)d * H + h], encf(a));
}

__global__ void edge_denom_k(const int* __restrict__ src, const int* __restrict__ dst,
                             const float* __restrict__ als, int lds_,
                             const float* __restrict__ ald, int ldd,
                             const unsigned* __restrict__ amax, float* __restrict__ denom,
                             int E, int H) {
  int i = blockIdx.x * blockDim.x + threadIdx.x;
  if (i >= E * H) return;
  int e = i / H, h = i - e * H;
  int s = src[e], d = dst[e];
  float a = edge_score(als, lds_, ald, ldd, s, d, h);
  float ex = __expf(a - decf(amax[(size_t)d * H + h]));
  atomicAdd(&denom[(size_t)d * H + h], ex);
}

__global__ void edge_scatter_k(const int* __restrict__ src, const int* __restrict__ dst,
                               const float* __restrict__ als, int lds_,
                               const float* __restrict__ ald, int ldd,
                               const unsigned* __restrict__ amax,
                               const float* __restrict__ denom,
                               const float* __restrict__ xs, int ldx,
                               float* __restrict__ out, int ldo, int colbase,
                               int E, int H, int C) {
  int i = blockIdx.x * blockDim.x + threadIdx.x;
  if (i >= E * H) return;
  int e = i / H, h = i - e * H;
  int s = src[e], d = dst[e];
  float a = edge_score(als, lds_, ald, ldd, s, d, h);
  float alpha = __expf(a - decf(amax[(size_t)d * H + h])) /
                (denom[(size_t)d * H + h] + 1e-16f);
  const float* xr = xs + (size_t)s * ldx + h * C;
  float* orow = out + (size_t)d * ldo + colbase + h * C;
  for (int c = 0; c < C; ++c) atomicAdd(&orow[c], xr[c] * alpha);
}

__global__ void elu_k(float* p, size_t n) {
  size_t i = (size_t)blockIdx.x * blockDim.x + threadIdx.x;
  size_t st = (size_t)gridDim.x * blockDim.x;
  for (; i < n; i += st) {
    float v = p[i];
    p[i] = (v > 0.f) ? v : expm1f(v);
  }
}

__global__ void lsm_k(float* p, int N, int C) {
  int n = blockIdx.x * blockDim.x + threadIdx.x;
  if (n >= N) return;
  float* r = p + (size_t)n * C;
  float m = -3.4e38f;
  for (int c = 0; c < C; ++c) m = fmaxf(m, r[c]);
  float s = 0.f;
  for (int c = 0; c < C; ++c) s += __expf(r[c] - m);
  float l = logf(s);
  for (int c = 0; c < C; ++c) r[c] = r[c] - m - l;
}

// ---------------------------------------------------------------------------

extern "C" void kernel_launch(void* const* d_in, const int* in_sizes, int n_in,
                              void* d_out, int out_size, void* d_ws, size_t ws_size,
                              hipStream_t stream) {
  const float* x    = (const float*)d_in[0];
  const int*   e1   = (const int*)d_in[1];
  const int*   e2   = (const int*)d_in[2];
  const float* Ws1  = (const float*)d_in[3];
  const float* Wd1  = (const float*)d_in[4];
  const float* Wd21 = (const float*)d_in[5];
  const float* as1  = (const float*)d_in[6];
  const float* ad1  = (const float*)d_in[7];
  const float* ad21 = (const float*)d_in[8];
  const float* Ws2  = (const float*)d_in[9];
  const float* Wd2  = (const float*)d_in[10];
  const float* Wd22 = (const float*)d_in[11];
  const float* as2  = (const float*)d_in[12];
  const float* ad2  = (const float*)d_in[13];
  const float* ad22 = (const float*)d_in[14];

  const int F_IN = 500, H1 = 4, C1 = 32, C2 = 40;
  const int HID = 2 * H1 * C1;                 // 256
  const int N  = in_sizes[0] / F_IN;           // 50000
  const int E1 = in_sizes[1] / 2;
  const int E2 = in_sizes[2] / 2;
  const int* s1 = e1;  const int* d1 = e1 + E1;
  const int* s2 = e2;  const int* d2 = e2 + E2;

  const int NCH1 = (F_IN + 31) / 32;           // 16 K-chunks, layer 1
  const int NCH2 = HID / 32;                   // 8  K-chunks, layer 2
  const int NT1 = 8, NT2 = 3;

  // ---- workspace carve (float units, 256B aligned regions) ----
  char* base = (char*)d_ws;
  size_t off = 0;
  auto carve = [&](size_t nf) -> float* {
    float* p = (float*)(base + off);
    off += ((nf * sizeof(float) + 255) / 256) * 256;
    return p;
  };
  float*    hbuf  = carve((size_t)N * HID);      // layer1 out / layer2 in
  float*    xs1   = carve((size_t)N * 128);      // x @ W_src1
  float*    wd1   = carve((size_t)F_IN * 8);     // collapsed dst weights L1
  float*    als1  = carve((size_t)N * H1);
  float*    ald8  = carve((size_t)N * 8);        // [:,0:4]=al_d1 [:,4:8]=al_d21
  unsigned* amax  = (unsigned*)carve((size_t)N * H1);
  float*    denom = carve((size_t)N * H1);
  float*    xs2   = carve((size_t)N * C2);
  float*    wd2   = carve((size_t)HID * 2);
  float*    als2  = carve((size_t)N);
  float*    al2   = carve((size_t)N * 2);
  _Float16* bp1   = (_Float16*)carve((size_t)NCH1 * NT1 * 512 / 2);  // f16 packed W_src1
  _Float16* bp2   = (_Float16*)carve((size_t)NCH2 * NT2 * 512 / 2);  // f16 packed W_src2
  (void)ws_size; (void)n_in; (void)out_size;

  const int TB = 256;
  const int FILLG = 2048;
  auto cdiv = [](long a, long b) { return (int)((a + b - 1) / b); };

  // ================= Layer 1 =================
  collapse_k<<<cdiv(F_IN * H1, TB), TB, 0, stream>>>(Wd1,  ad1,  wd1, F_IN, H1, C1, 8, 0);
  collapse_k<<<cdiv(F_IN * H1, TB), TB, 0, stream>>>(Wd21, ad21, wd1, F_IN, H1, C1, 8, 4);

  pack_b_k<<<cdiv((long)NCH1 * NT1 * 512, TB), TB, 0, stream>>>(Ws1, bp1, F_IN, 128, NT1, NCH1);
  wmma_gemm_k<NT1><<<cdiv(N, 16), 32, 0, stream>>>(x, bp1, xs1, N, F_IN, 128, 128, NCH1);

  als_k<<<cdiv((long)N * H1, TB), TB, 0, stream>>>(xs1, as1, als1, N, H1, C1);
  dotal_k<<<cdiv(N, TB), TB, 0, stream>>>(x, wd1, ald8, N, F_IN, 8);

  fill_f<<<FILLG, TB, 0, stream>>>(hbuf, 0.f, (size_t)N * HID);

  // ---- 1-hop ----
  fill_u<<<FILLG, TB, 0, stream>>>(amax, 0u, (size_t)N * H1);
  fill_f<<<FILLG, TB, 0, stream>>>(denom, 0.f, (size_t)N * H1);
  edge_amax_k  <<<cdiv((long)E1 * H1, TB), TB, 0, stream>>>(s1, d1, als1, H1, ald8, 8, amax, E1, H1);
  edge_denom_k <<<cdiv((long)E1 * H1, TB), TB, 0, stream>>>(s1, d1, als1, H1, ald8, 8, amax, denom, E1, H1);
  edge_scatter_k<<<cdiv((long)E1 * H1, TB), TB, 0, stream>>>(s1, d1, als1, H1, ald8, 8, amax, denom,
                                                             xs1, 128, hbuf, HID, 0, E1, H1, C1);
  // ---- 2-hop ----
  fill_u<<<FILLG, TB, 0, stream>>>(amax, 0u, (size_t)N * H1);
  fill_f<<<FILLG, TB, 0, stream>>>(denom, 0.f, (size_t)N * H1);
  edge_amax_k  <<<cdiv((long)E2 * H1, TB), TB, 0, stream>>>(s2, d2, als1, H1, ald8 + 4, 8, amax, E2, H1);
  edge_denom_k <<<cdiv((long)E2 * H1, TB), TB, 0, stream>>>(s2, d2, als1, H1, ald8 + 4, 8, amax, denom, E2, H1);
  edge_scatter_k<<<cdiv((long)E2 * H1, TB), TB, 0, stream>>>(s2, d2, als1, H1, ald8 + 4, 8, amax, denom,
                                                             xs1, 128, hbuf, HID, 128, E2, H1, C1);

  elu_k<<<FILLG, TB, 0, stream>>>(hbuf, (size_t)N * HID);

  // ================= Layer 2 =================
  collapse_k<<<cdiv(HID, TB), TB, 0, stream>>>(Wd2,  ad2,  wd2, HID, 1, C2, 2, 0);
  collapse_k<<<cdiv(HID, TB), TB, 0, stream>>>(Wd22, ad22, wd2, HID, 1, C2, 2, 1);

  pack_b_k<<<cdiv((long)NCH2 * NT2 * 512, TB), TB, 0, stream>>>(Ws2, bp2, HID, C2, NT2, NCH2);
  wmma_gemm_k<NT2><<<cdiv(N, 16), 32, 0, stream>>>(hbuf, bp2, xs2, N, HID, C2, C2, NCH2);

  als_k<<<cdiv(N, TB), TB, 0, stream>>>(xs2, as2, als2, N, 1, C2);
  dotal_k<<<cdiv(N, TB), TB, 0, stream>>>(hbuf, wd2, al2, N, HID, 2);

  float* outp = (float*)d_out;
  fill_f<<<FILLG, TB, 0, stream>>>(outp, 0.f, (size_t)N * 2 * C2);

  // ---- 1-hop ----
  fill_u<<<FILLG, TB, 0, stream>>>(amax, 0u, (size_t)N);
  fill_f<<<FILLG, TB, 0, stream>>>(denom, 0.f, (size_t)N);
  edge_amax_k  <<<cdiv(E1, TB), TB, 0, stream>>>(s1, d1, als2, 1, al2, 2, amax, E1, 1);
  edge_denom_k <<<cdiv(E1, TB), TB, 0, stream>>>(s1, d1, als2, 1, al2, 2, amax, denom, E1, 1);
  edge_scatter_k<<<cdiv(E1, TB), TB, 0, stream>>>(s1, d1, als2, 1, al2, 2, amax, denom,
                                                  xs2, C2, outp, 2 * C2, 0, E1, 1, C2);
  // ---- 2-hop ----
  fill_u<<<FILLG, TB, 0, stream>>>(amax, 0u, (size_t)N);
  fill_f<<<FILLG, TB, 0, stream>>>(denom, 0.f, (size_t)N);
  edge_amax_k  <<<cdiv(E2, TB), TB, 0, stream>>>(s2, d2, als2, 1, al2 + 1, 2, amax, E2, 1);
  edge_denom_k <<<cdiv(E2, TB), TB, 0, stream>>>(s2, d2, als2, 1, al2 + 1, 2, amax, denom, E2, 1);
  edge_scatter_k<<<cdiv(E2, TB), TB, 0, stream>>>(s2, d2, als2, 1, al2 + 1, 2, amax, denom,
                                                  xs2, C2, outp, 2 * C2, C2, E2, 1, C2);

  lsm_k<<<cdiv(N, TB), TB, 0, stream>>>(outp, N, 2 * C2);
}
